// NaiveMHLA_7868380086609
// MI455X (gfx1250) — matrix operand order repaired and checked by reading, not compiled, not run on hardware
//
#include <hip/hip_runtime.h>

// ---------------------------------------------------------------------------
// MLA (latent attention) for gfx1250: all GEMMs via v_wmma_f32_16x16x32_f16.
// Convention: every GEMM is "TN": C[m,n] = sum_k A[m,k] * Bt[n,k]
//   A  : row-major M x K f16
//   Bt : row-major N x K f16 (i.e. B stored transposed)
// ---------------------------------------------------------------------------

typedef __attribute__((ext_vector_type(16))) _Float16 v16h;
typedef __attribute__((ext_vector_type(8)))  _Float16 v8h;
typedef __attribute__((ext_vector_type(8)))  float    v8f;

#define WMMA_F16(a, b, c) \
  __builtin_amdgcn_wmma_f32_16x16x32_f16(false, (a), false, (b), (short)0, (c), false, false)

#define B_   4
#define T_   2048
#define C_   1024
#define QL_  256
#define KVL_ 256
#define NH_  16
#define HS_  64

// Load 16 f16 as two 16B chunks at element offsets o1 / o2.
__device__ __forceinline__ v16h ld16pair(const _Float16* p, int o1, int o2) {
  v8h lo = *(const v8h*)(p + o1);
  v8h hi = *(const v8h*)(p + o2);
  v16h r;
#pragma unroll
  for (int i = 0; i < 8; ++i) { r[i] = lo[i]; r[i + 8] = hi[i]; }
  return r;
}
// A fragment 16x32: rowp = &A[(m0 + (lane&15))*lda + k0]
__device__ __forceinline__ v16h frag_a(const _Float16* rowp, int sub) {
  const int o = sub ? 8 : 0;
  return ld16pair(rowp, o, o + 16);
}
// B fragment 32x16 (from Bt): rowp = &Bt[(n0 + (lane&15))*ldb + k0]
__device__ __forceinline__ v16h frag_b(const _Float16* rowp, int sub) {
  const int o = sub ? 8 : 0;
  return ld16pair(rowp, o, o + 16);
}

// ---------------------------------------------------------------------------
// f32 -> f16 converts (straight + transposed)
// ---------------------------------------------------------------------------
__global__ void k_cvt(_Float16* __restrict__ dst, const float* __restrict__ src, int n) {
  int i = blockIdx.x * 256 + threadIdx.x;
  if (i < n) dst[i] = (_Float16)src[i];
}
// src is (R x Ccol) row-major f32; dst is (Ccol x R) row-major f16.
__global__ void k_cvt_t(_Float16* __restrict__ dst, const float* __restrict__ src,
                        int R, int Ccol) {
  int i = blockIdx.x * 256 + threadIdx.x;
  if (i < R * Ccol) {
    int r = i / Ccol, c = i % Ccol;
    dst[(size_t)c * R + r] = (_Float16)src[i];
  }
}

// ---------------------------------------------------------------------------
// Generic TN GEMM, 8 waves/block, 32x32 output tile per wave (2x2 WMMA).
// Optional f32 / f16 / f16-transposed outputs.
// ---------------------------------------------------------------------------
__global__ __launch_bounds__(256) void k_gemm(
    const _Float16* __restrict__ A, int lda,
    const _Float16* __restrict__ Bt, int ldb,
    float* __restrict__ Cf, int ldcf,
    _Float16* __restrict__ Ch, int ldch,
    _Float16* __restrict__ ChT, int ldct,
    int M, int N, int K) {
  const int wave = threadIdx.x >> 5;
  const int lane = threadIdx.x & 31;
  const int sub  = lane >> 4;      // which 16-lane half
  const int l15  = lane & 15;

  const int tilesN = N >> 5;
  const int tiles  = (M >> 5) * tilesN;
  const int tile   = blockIdx.x * 8 + wave;
  if (tile >= tiles) return;
  const int m0 = (tile / tilesN) * 32;
  const int n0 = (tile % tilesN) * 32;

  const _Float16* a0p = A + (size_t)(m0 + l15) * lda;
  const _Float16* a1p = a0p + (size_t)16 * lda;
  const _Float16* b0p = Bt + (size_t)(n0 + l15) * ldb;
  const _Float16* b1p = b0p + (size_t)16 * ldb;

  v8f acc00 = {}, acc01 = {}, acc10 = {}, acc11 = {};
  for (int k0 = 0; k0 < K; k0 += 32) {
    v16h a0 = frag_a(a0p + k0, sub);
    v16h a1 = frag_a(a1p + k0, sub);
    v16h b0 = frag_b(b0p + k0, sub);
    v16h b1 = frag_b(b1p + k0, sub);
    acc00 = WMMA_F16(a0, b0, acc00);
    acc01 = WMMA_F16(a0, b1, acc01);
    acc10 = WMMA_F16(a1, b0, acc10);
    acc11 = WMMA_F16(a1, b1, acc11);
  }
#pragma unroll
  for (int r = 0; r < 8; ++r) {
    const int ml = m0 + r + sub * 8;
    const int nl = n0 + l15;
    const float v00 = acc00[r], v01 = acc01[r], v10 = acc10[r], v11 = acc11[r];
    if (Cf) {
      Cf[(size_t)ml * ldcf + nl]            = v00;
      Cf[(size_t)ml * ldcf + nl + 16]       = v01;
      Cf[(size_t)(ml + 16) * ldcf + nl]     = v10;
      Cf[(size_t)(ml + 16) * ldcf + nl + 16]= v11;
    }
    if (Ch) {
      Ch[(size_t)ml * ldch + nl]             = (_Float16)v00;
      Ch[(size_t)ml * ldch + nl + 16]        = (_Float16)v01;
      Ch[(size_t)(ml + 16) * ldch + nl]      = (_Float16)v10;
      Ch[(size_t)(ml + 16) * ldch + nl + 16] = (_Float16)v11;
    }
    if (ChT) {
      ChT[(size_t)nl * ldct + ml]              = (_Float16)v00;
      ChT[(size_t)(nl + 16) * ldct + ml]       = (_Float16)v01;
      ChT[(size_t)nl * ldct + ml + 16]         = (_Float16)v10;
      ChT[(size_t)(nl + 16) * ldct + ml + 16]  = (_Float16)v11;
    }
  }
}

// ---------------------------------------------------------------------------
// Flash-style latent attention. Grid: (T/64, NH, B). Block: 128 (4 waves).
// Each wave: 16 query rows; online softmax over key blocks of 32;
// ctx accumulator 16x256 fp32 in VGPRs (16 x v8f).
// ---------------------------------------------------------------------------
__global__ __launch_bounds__(128) void k_attn(
    const _Float16* __restrict__ qlat,   // (B*T, NH*KVL) ld 4096
    const _Float16* __restrict__ ckv,    // (B*T, KVL)
    const _Float16* __restrict__ ckvT,   // (KVL, B*T) ld 8192
    const _Float16* __restrict__ veffT,  // (C, KVL): row h*64+d = v_eff[h][:,d]
    float* __restrict__ y) {             // (B*T, C)
  __shared__ __align__(16) _Float16 ldsP[4][16 * 32];
  __shared__ __align__(16) _Float16 ldsC[4][16 * 256];

  const int qblk = blockIdx.x, h = blockIdx.y, b = blockIdx.z;
  const int wave = threadIdx.x >> 5, lane = threadIdx.x & 31;
  const int sub = lane >> 4, l15 = lane & 15;
  const int q0 = qblk * 64 + wave * 16;

  const _Float16* qrow = qlat + (size_t)(b * T_ + q0 + l15) * (NH_ * KVL_) + h * KVL_;

  v8f ctx[16];
#pragma unroll
  for (int t = 0; t < 16; ++t) ctx[t] = {};
  float mrow[8], lrow[8];
#pragma unroll
  for (int r = 0; r < 8; ++r) { mrow[r] = -1e30f; lrow[r] = 0.0f; }

  const int kend = qblk * 64 + 64;   // uniform across the workgroup (causal)
  for (int kb = 0; kb < kend; kb += 32) {
    // ---- scores S = Q(16x256) @ Kc^T(256x32) : two 16x16 tiles -------------
    v8f s0 = {}, s1 = {};
    const _Float16* k0row = ckv + (size_t)(b * T_ + kb + l15) * KVL_;
    const _Float16* k1row = k0row + (size_t)16 * KVL_;
#pragma unroll
    for (int kk = 0; kk < 8; ++kk) {
      v16h qa = frag_a(qrow + kk * 32, sub);
      v16h b0 = frag_b(k0row + kk * 32, sub);
      v16h b1 = frag_b(k1row + kk * 32, sub);
      s0 = WMMA_F16(qa, b0, s0);
      s1 = WMMA_F16(qa, b1, s1);
    }
    // ---- online softmax (per row r+8*sub; row spans 16 lanes of this half) -
#pragma unroll
    for (int r = 0; r < 8; ++r) {
      const int m = q0 + r + sub * 8;
      float e0 = s0[r] * 0.125f;                 // 1/sqrt(64)
      float e1 = s1[r] * 0.125f;
      if (kb + l15 > m)      e0 = -1e30f;        // causal mask
      if (kb + 16 + l15 > m) e1 = -1e30f;
      float mx = fmaxf(e0, e1);
#pragma unroll
      for (int d = 1; d < 16; d <<= 1) mx = fmaxf(mx, __shfl_xor(mx, d));
      const float nm = fmaxf(mrow[r], mx);
      const float alpha = __expf(mrow[r] - nm);
      mrow[r] = nm;
      float p0 = __expf(e0 - nm), p1 = __expf(e1 - nm);
      float ps = p0 + p1;
#pragma unroll
      for (int d = 1; d < 16; d <<= 1) ps += __shfl_xor(ps, d);
      lrow[r] = lrow[r] * alpha + ps;
#pragma unroll
      for (int t = 0; t < 16; ++t) ctx[t][r] *= alpha;
      ldsP[wave][(r + sub * 8) * 32 + l15]      = (_Float16)p0;
      ldsP[wave][(r + sub * 8) * 32 + 16 + l15] = (_Float16)p1;
    }
    __syncthreads();
    // ---- ctx += P(16x32) @ Kc(32x256) --------------------------------------
    v16h pa = frag_a(&ldsP[wave][l15 * 32], sub);
    const _Float16* kT = ckvT + (size_t)l15 * (B_ * T_) + (size_t)b * T_ + kb;
#pragma unroll
    for (int t = 0; t < 16; ++t) {
      v16h bb = frag_b(kT + (size_t)t * 16 * (B_ * T_), sub);
      ctx[t] = WMMA_F16(pa, bb, ctx[t]);
    }
    __syncthreads();
  }

  // ---- normalize, spill ctx to LDS as f16 A-operand ------------------------
#pragma unroll
  for (int r = 0; r < 8; ++r) {
    const float inv = 1.0f / lrow[r];
#pragma unroll
    for (int t = 0; t < 16; ++t)
      ldsC[wave][(r + sub * 8) * 256 + t * 16 + l15] = (_Float16)(ctx[t][r] * inv);
  }
  __syncthreads();

  // ---- y_head = ctx(16x256) @ v_eff[h](256x64) -----------------------------
  v8f yac[4];
#pragma unroll
  for (int j = 0; j < 4; ++j) yac[j] = {};
  const _Float16* crow = &ldsC[wave][l15 * 256];
  const _Float16* vrow = veffT + (size_t)(h * HS_ + l15) * KVL_;
#pragma unroll
  for (int kk = 0; kk < 8; ++kk) {
    v16h ca = frag_a(crow + kk * 32, sub);
#pragma unroll
    for (int j = 0; j < 4; ++j) {
      v16h vb = frag_b(vrow + (size_t)j * 16 * KVL_ + kk * 32, sub);
      yac[j] = WMMA_F16(ca, vb, yac[j]);
    }
  }
#pragma unroll
  for (int r = 0; r < 8; ++r) {
    const size_t row = (size_t)(b * T_ + q0 + r + sub * 8) * C_ + h * HS_;
#pragma unroll
    for (int j = 0; j < 4; ++j) y[row + j * 16 + l15] = yac[j][r];
  }
}

// ---------------------------------------------------------------------------
// Host orchestration
// ---------------------------------------------------------------------------
static inline int gemm_blocks(int M, int N) { return ((M / 32) * (N / 32) + 7) / 8; }

extern "C" void kernel_launch(void* const* d_in, const int* in_sizes, int n_in,
                              void* d_out, int out_size, void* d_ws, size_t ws_size,
                              hipStream_t stream) {
  (void)in_sizes; (void)n_in; (void)out_size; (void)ws_size;
  const float* x    = (const float*)d_in[0];  // (B,T,C)
  const float* Wdq  = (const float*)d_in[1];  // (QL,C)
  const float* Wuq  = (const float*)d_in[2];  // (C,QL)
  const float* Wdkv = (const float*)d_in[3];  // (KVL,C)
  const float* Wuk  = (const float*)d_in[4];  // (C,KVL)
  const float* Wuv  = (const float*)d_in[5];  // (C,KVL)
  const float* Wo   = (const float*)d_in[6];  // (C,C)

  float* y_out   = (float*)d_out;
  float* ckv_out = y_out + (size_t)B_ * T_ * C_;

  const size_t BT = (size_t)B_ * T_;  // 8192
  unsigned char* wp = (unsigned char*)d_ws;
  auto take = [&](size_t elems) -> _Float16* {
    _Float16* p = (_Float16*)wp;
    wp += ((elems * sizeof(_Float16)) + 255) & ~(size_t)255;
    return p;
  };
  _Float16* x_h    = take(BT * C_);        // x, row-major
  _Float16* Wdkv_h = take(KVL_ * C_);      // = Bt for c_kv
  _Float16* Wdq_h  = take(QL_ * C_);       // = Bt for q_low
  _Float16* Wuq_h  = take(C_ * QL_);       // = Bt for q
  _Float16* Wo_h   = take(C_ * C_);        // A for veffT
  _Float16* WuqT_h = take(QL_ * C_);       // Wuq^T  (256 x 1024)
  _Float16* WukT_h = take(KVL_ * C_);      // Wuk^T  (256 x 1024)
  _Float16* WdqT_h = take(C_ * QL_);       // Wdq^T  (1024 x 256)
  _Float16* WuvT_h = take(KVL_ * C_);      // Wuv^T  (256 x 1024)
  _Float16* tmpT_h = take(KVL_ * QL_);     // (Wuq^T @ Wuk)^T = Wuk^T @ Wuq
  _Float16* keffT_h= take(KVL_ * C_);      // k_eff^T (256 x 1024)
  _Float16* veffT_h= take(C_ * KVL_);      // (Wuv^T @ Wo^T)^T = Wo @ Wuv
  _Float16* ckv_h  = take(BT * KVL_);      // c_kv f16
  _Float16* ckvT_h = take(KVL_ * BT);      // c_kv^T f16 (256 x 8192)
  _Float16* qlow_h = take(BT * QL_);       // x @ Wdq^T
  _Float16* q_h    = take(BT * C_);        // q_low @ Wuq^T
  _Float16* qlat_h = take(BT * NH_ * KVL_);// (B*T, NH*256)

  // --- converts ---
  k_cvt<<<(int)((BT * C_ + 255) / 256), 256, 0, stream>>>(x_h, x, (int)(BT * C_));
  k_cvt<<<(KVL_ * C_ + 255) / 256, 256, 0, stream>>>(Wdkv_h, Wdkv, KVL_ * C_);
  k_cvt<<<(QL_ * C_ + 255) / 256, 256, 0, stream>>>(Wdq_h, Wdq, QL_ * C_);
  k_cvt<<<(C_ * QL_ + 255) / 256, 256, 0, stream>>>(Wuq_h, Wuq, C_ * QL_);
  k_cvt<<<(C_ * C_ + 255) / 256, 256, 0, stream>>>(Wo_h, Wo, C_ * C_);
  k_cvt_t<<<(C_ * QL_ + 255) / 256, 256, 0, stream>>>(WuqT_h, Wuq, C_, QL_);
  k_cvt_t<<<(C_ * KVL_ + 255) / 256, 256, 0, stream>>>(WukT_h, Wuk, C_, KVL_);
  k_cvt_t<<<(QL_ * C_ + 255) / 256, 256, 0, stream>>>(WdqT_h, Wdq, QL_, C_);
  k_cvt_t<<<(C_ * KVL_ + 255) / 256, 256, 0, stream>>>(WuvT_h, Wuv, C_, KVL_);

  // --- absorbed weights ---
  // tmpT = Wuk^T @ Wuq                       (256 x 256), K=1024
  k_gemm<<<gemm_blocks(KVL_, QL_), 256, 0, stream>>>(
      WukT_h, C_, WuqT_h, C_, nullptr, 0, tmpT_h, QL_, nullptr, 0, KVL_, QL_, C_);
  // keffT = tmpT @ Wdq                       (256 x 1024), K=256
  k_gemm<<<gemm_blocks(KVL_, C_), 256, 0, stream>>>(
      tmpT_h, QL_, WdqT_h, QL_, nullptr, 0, keffT_h, C_, nullptr, 0, KVL_, C_, QL_);
  // veffT = Wo @ Wuv                         (1024 x 256), K=1024
  k_gemm<<<gemm_blocks(C_, KVL_), 256, 0, stream>>>(
      Wo_h, C_, WuvT_h, C_, nullptr, 0, veffT_h, KVL_, nullptr, 0, C_, KVL_, C_);

  // --- projections ---
  // c_kv = x @ Wdkv^T  -> f32 out + f16 + f16-transposed
  k_gemm<<<gemm_blocks((int)BT, KVL_), 256, 0, stream>>>(
      x_h, C_, Wdkv_h, C_, ckv_out, KVL_, ckv_h, KVL_, ckvT_h, (int)BT,
      (int)BT, KVL_, C_);
  // q_low = x @ Wdq^T
  k_gemm<<<gemm_blocks((int)BT, QL_), 256, 0, stream>>>(
      x_h, C_, Wdq_h, C_, nullptr, 0, qlow_h, QL_, nullptr, 0, (int)BT, QL_, C_);
  // q = q_low @ Wuq^T
  k_gemm<<<gemm_blocks((int)BT, C_), 256, 0, stream>>>(
      qlow_h, QL_, Wuq_h, QL_, nullptr, 0, q_h, C_, nullptr, 0, (int)BT, C_, QL_);
  // q_lat[:, h*256:(h+1)*256] = q[:, h*64:(h+1)*64] @ k_eff[h]   (K = 64)
  for (int h = 0; h < NH_; ++h) {
    k_gemm<<<gemm_blocks((int)BT, KVL_), 256, 0, stream>>>(
        q_h + h * HS_, C_, keffT_h + h * HS_, C_,
        nullptr, 0, qlat_h + h * KVL_, NH_ * KVL_, nullptr, 0,
        (int)BT, KVL_, HS_);
  }

  // --- flash attention + absorbed output projection ---
  dim3 grid(T_ / 64, NH_, B_);
  k_attn<<<grid, 128, 0, stream>>>(qlat_h, ckv_h, ckvT_h, veffT_h, y_out);
}
